// NeighborAttention_51170240365036
// MI455X (gfx1250) — compile-verified, matrix operand order checked by reading, not run on hardware
//
#include <hip/hip_runtime.h>
#include <hip/hip_bf16.h>
#include <math.h>

// ---------------- problem constants ----------------
#define E_TOT 262144
#define N_TOT 8192
// H=128, NV=16, NH=4, D=32

typedef __attribute__((ext_vector_type(16))) __bf16 bf16x16;
typedef __attribute__((ext_vector_type(8)))  float  f32x8;

// ---------------- packed-weight fragment offsets (bf16 halfs) ----------------
#define PK_WRED 0         // 320x256 : 10 kt x 16 nt
#define PK_WB1  81920     // 384x128 : 12 x 8
#define PK_WB2  131072    // 128x128 : 4 x 8
#define PK_WB3  147456    // 128x8(pad16) : 4 x 1
#define PK_WV1  149504    // 256x128 : 8 x 8
#define PK_WV2  182272    // 128x128 : 4 x 8
#define PK_WV3  198656    // 128x128 : 4 x 8
#define PK_WO   215040    // 128x128 : 4 x 8
#define PK_FRAGS 452

// ---------------- workspace layout (bytes) ----------------
#define OFF_PACK ((size_t)0)                               // 462848 B bf16
#define OFF_VDT  ((size_t)462848)                          // E*48 f32
#define OFF_LOG  (OFF_VDT  + (size_t)E_TOT*48*4)           // E*8  f32 (logits -> exp in place)
#define OFF_V    (OFF_LOG  + (size_t)E_TOT*8*4)            // E*128 f32
#define OFF_NMAX (OFF_V    + (size_t)E_TOT*128*4)          // N*8 u32 (ordered-float max)
#define OFF_NSUM (OFF_NMAX + (size_t)N_TOT*8*4)            // N*8 f32
#define OFF_HAGG (OFF_NSUM + (size_t)N_TOT*8*4)            // N*128 f32
#define OFF_VAGG (OFF_HAGG + (size_t)N_TOT*128*4)          // N*48 f32
#define ZCLEAR_BYTES ((size_t)N_TOT*8*4*2 + (size_t)N_TOT*128*4 + (size_t)N_TOT*48*4)

// ---------------- helpers ----------------
__device__ __forceinline__ unsigned short f2bf(float f) {
  unsigned int u = __float_as_uint(f);
  unsigned int r = u + 0x7FFFu + ((u >> 16) & 1u);   // round-to-nearest-even
  return (unsigned short)(r >> 16);
}
__device__ __forceinline__ unsigned int f2ord(float f) {  // order-preserving float->uint
  unsigned int u = __float_as_uint(f);
  return (u & 0x80000000u) ? ~u : (u | 0x80000000u);
}
__device__ __forceinline__ float ord2f(unsigned int u) {
  unsigned int b = (u & 0x80000000u) ? (u ^ 0x80000000u) : ~u;
  return __uint_as_float(b);
}
__device__ __forceinline__ float gelu_t(float x) {
  float x3 = x * x * x;
  return 0.5f * x * (1.0f + tanhf(0.7978845608028654f * (x + 0.044715f * x3)));
}
__device__ __forceinline__ f32x8 wmma_bf16(bf16x16 a, bf16x16 b, f32x8 c) {
  return __builtin_amdgcn_wmma_f32_16x16x32_bf16(false, a, false, b, (short)0, c, false, false);
}
// A fragment from LDS row-major bf16 tile: lane<16 -> K {0..7,16..23}, lane>=16 -> +8
__device__ __forceinline__ bf16x16 ld_afrag(const unsigned short* buf, int stride, int lane, int kOff) {
  int row = lane & 15;
  int kb  = kOff + ((lane & 16) ? 8 : 0);
  const unsigned short* p = buf + row * stride + kb;
  union { uint4 u[2]; bf16x16 v; } t;
  t.u[0] = *(const uint4*)p;
  t.u[1] = *(const uint4*)(p + 16);   // K gap of 16
  return t.v;
}
// B fragment from packed global weights: 32B contiguous per lane
__device__ __forceinline__ bf16x16 ld_bfrag(const unsigned short* pk, int frag, int lane) {
  const unsigned short* p = pk + (((size_t)frag * 32 + lane) << 4);
  union { uint4 u[2]; bf16x16 v; } t;
  t.u[0] = *(const uint4*)p;
  t.u[1] = *(const uint4*)(p + 8);
  return t.v;
}
// C tile (f32 regs) -> LDS bf16 row-major
__device__ __forceinline__ void st_cfrag(unsigned short* buf, int stride, int lane, int nOff, f32x8 c) {
  int n  = nOff + (lane & 15);
  int mb = (lane & 16) ? 8 : 0;
#pragma unroll
  for (int j = 0; j < 8; ++j) buf[(mb + j) * stride + n] = f2bf(c[j]);
}
#define ZERO8 {0.f,0.f,0.f,0.f,0.f,0.f,0.f,0.f}

// ================= kernel 0: pack weights to bf16 WMMA-B fragments =================
__global__ void __launch_bounds__(256) pack_kernel(
    const float* Wred, const float* Wb1, const float* Wb2, const float* Wb3,
    const float* Wv1, const float* Wv2, const float* Wv3, const float* WO,
    unsigned short* pk) {
  int t = blockIdx.x * blockDim.x + threadIdx.x;
  int f = t >> 5, lane = t & 31;
  if (f >= PK_FRAGS) return;
  const int fragStart[9] = {0, 160, 256, 288, 292, 356, 388, 420, 452};
  int m = 0;
  while (f >= fragStart[m + 1]) ++m;
  int fl = f - fragStart[m];
  const float* src; int Nn, ntiles, pkoff;
  switch (m) {
    case 0: src = Wred; Nn = 256; ntiles = 16; pkoff = PK_WRED; break;
    case 1: src = Wb1;  Nn = 128; ntiles = 8;  pkoff = PK_WB1;  break;
    case 2: src = Wb2;  Nn = 128; ntiles = 8;  pkoff = PK_WB2;  break;
    case 3: src = Wb3;  Nn = 8;   ntiles = 1;  pkoff = PK_WB3;  break;
    case 4: src = Wv1;  Nn = 128; ntiles = 8;  pkoff = PK_WV1;  break;
    case 5: src = Wv2;  Nn = 128; ntiles = 8;  pkoff = PK_WV2;  break;
    case 6: src = Wv3;  Nn = 128; ntiles = 8;  pkoff = PK_WV3;  break;
    default:src = WO;   Nn = 128; ntiles = 8;  pkoff = PK_WO;   break;
  }
  int kt = fl / ntiles, nt = fl % ntiles;
  int n  = nt * 16 + (lane & 15);
  int kb = kt * 32 + ((lane & 16) ? 8 : 0);
  unsigned short* dst = pk + pkoff + (((size_t)fl * 32 + lane) << 4);
#pragma unroll
  for (int j = 0; j < 16; ++j) {
    int k = kb + ((j < 8) ? j : (16 + (j - 8)));
    float v = (n < Nn) ? src[(size_t)k * Nn + n] : 0.0f;
    dst[j] = f2bf(v);
  }
}

// ================= kernel 1: per-edge fused WMMA chain =================
#define S1 328   // A1 stride (halfs): feat_s(64)+h_E(256)=320 cols
#define S2 392   // A2 stride: h_V(128)+hE(256)=384 cols
#define SW 136   // w/V stride: 128 cols
__global__ void __launch_bounds__(64) edge_kernel(
    const float* __restrict__ hV, const float* __restrict__ hE_g,
    const int* __restrict__ center, const int* __restrict__ dst,
    const float* __restrict__ frameR, const float* __restrict__ frameT,
    const float* __restrict__ hVvec, const float* __restrict__ Wevf,
    const float* __restrict__ bnG, const float* __restrict__ bnB,
    const float* __restrict__ bnM, const float* __restrict__ bnVr,
    const float* __restrict__ bb1, const float* __restrict__ bb2, const float* __restrict__ bb3,
    const float* __restrict__ bv1, const float* __restrict__ bv2, const float* __restrict__ bv3,
    const unsigned short* __restrict__ pk,
    float* __restrict__ wsVdt, float* __restrict__ wsLog, float* __restrict__ wsV,
    unsigned int* __restrict__ nodeMax, float* __restrict__ outFeat) {
  __shared__ alignas(16) unsigned short sA1[2][16 * S1];
  __shared__ alignas(16) unsigned short sA2[2][16 * S2];
  const int wave = threadIdx.x >> 5;
  const int lane = threadIdx.x & 31;
  const int tile = blockIdx.x * 2 + wave;
  const int e0 = tile * 16;
  unsigned short* A1 = sA1[wave];
  unsigned short* A2 = sA2[wave];
  unsigned short* W1 = A1;                 // 16 x SW, reuses A1 after stage 2
  unsigned short* W2 = A1 + 16 * SW;       // 16 x SW

  // prefetch next tile's h_E stream (gfx1250 global_prefetch)
  if (lane == 0 && e0 + 16 < E_TOT)
    __builtin_prefetch(hE_g + (size_t)(e0 + 16) * 256, 0, 1);

  // ---- stage A1 cols 64..319 : h_E rows (contiguous, coalesced) ----
  {
    int row = lane >> 1, half = lane & 1;
    const float* src = hE_g + (size_t)(e0 + row) * 256 + half * 128;
    unsigned short* dp = A1 + row * S1 + 64 + half * 128;
#pragma unroll
    for (int i = 0; i < 32; ++i) {
      float4 v = ((const float4*)src)[i];
      *(ushort4*)(dp + i * 4) = make_ushort4(f2bf(v.x), f2bf(v.y), f2bf(v.z), f2bf(v.w));
    }
  }
  // ---- stage A2 cols 0..127 : gathered h_V[center] ----
  {
    int row = lane >> 1, half = lane & 1;
    int c = center[e0 + row];
    const float* src = hV + (size_t)c * 128 + half * 64;
    unsigned short* dp = A2 + row * S2 + half * 64;
#pragma unroll
    for (int i = 0; i < 16; ++i) {
      float4 v = ((const float4*)src)[i];
      *(ushort4*)(dp + i * 4) = make_ushort4(f2bf(v.x), f2bf(v.y), f2bf(v.z), f2bf(v.w));
    }
  }
  // ---- geometry + feat_s (lanes 0..15, one edge each) ----
  if (lane < 16) {
    int e = e0 + lane;
    int c = center[e], d = dst[e];
    float R[9], t3[3];
#pragma unroll
    for (int i = 0; i < 9; ++i) R[i] = frameR[(size_t)e * 9 + i];
#pragma unroll
    for (int i = 0; i < 3; ++i) t3[i] = frameT[(size_t)e * 3 + i];
    const float* ps = hVvec + (size_t)c * 48;
    const float* pd = hVvec + (size_t)d * 48;
    float vsrc[48], vdt[48];
#pragma unroll
    for (int v = 0; v < 16; ++v) {
      float dx = pd[v * 3 + 0], dy = pd[v * 3 + 1], dz = pd[v * 3 + 2];
      vdt[v * 3 + 0] = R[0] * dx + R[1] * dy + R[2] * dz + t3[0];
      vdt[v * 3 + 1] = R[3] * dx + R[4] * dy + R[5] * dz + t3[1];
      vdt[v * 3 + 2] = R[6] * dx + R[7] * dy + R[8] * dz + t3[2];
      vsrc[v * 3 + 0] = ps[v * 3 + 0];
      vsrc[v * 3 + 1] = ps[v * 3 + 1];
      vsrc[v * 3 + 2] = ps[v * 3 + 2];
    }
    float4* wv = (float4*)(wsVdt + (size_t)e * 48);
#pragma unroll
    for (int i = 0; i < 12; ++i) wv[i] = ((float4*)vdt)[i];
    unsigned short* lrow = A1 + lane * S1;
    float* fo = outFeat + (size_t)e * 64;
#pragma unroll
    for (int o = 0; o < 16; ++o) {
      float sx = vdt[o * 3 + 0], sy = vdt[o * 3 + 1], sz = vdt[o * 3 + 2];
#pragma unroll
      for (int i = 0; i < 16; ++i) {
        float w0 = Wevf[o * 32 + i];
        float w1 = Wevf[o * 32 + 16 + i];
        sx += w0 * vdt[i * 3 + 0] + w1 * vsrc[i * 3 + 0];
        sy += w0 * vdt[i * 3 + 1] + w1 * vsrc[i * 3 + 1];
        sz += w0 * vdt[i * 3 + 2] + w1 * vsrc[i * 3 + 2];
      }
      float dist = sqrtf(sx * sx + sy * sy + sz * sz) + 1e-6f;
      float inv = 1.0f / dist;
      float a = sx * inv, b = sy * inv, g = sz * inv;
      float db = (dist - bnM[o]) * rsqrtf(bnVr[o] + 1e-5f) * bnG[o] + bnB[o];
      fo[o * 3 + 0] = a; fo[o * 3 + 1] = b; fo[o * 3 + 2] = g; fo[48 + o] = db;
      lrow[o * 3 + 0] = f2bf(a); lrow[o * 3 + 1] = f2bf(b); lrow[o * 3 + 2] = f2bf(g);
      lrow[48 + o] = f2bf(db);
    }
  }
  // LDS ops are in-order per wave (single wave owns its tile) -> no barrier needed.

  // ---- stage 2: hE = [feat_s, h_E] @ W_red  (K=320, N=256) -> A2 cols 128.. ----
#pragma unroll 1
  for (int nt = 0; nt < 16; ++nt) {
    f32x8 acc = ZERO8;
#pragma unroll
    for (int kt = 0; kt < 10; ++kt)
      acc = wmma_bf16(ld_afrag(A1, S1, lane, kt * 32), ld_bfrag(pk + PK_WRED, kt * 16 + nt, lane), acc);
    st_cfrag(A2, S2, lane, 128 + nt * 16, acc);
  }
  // ---- stage 4: w1 = relu([h_V, hE] @ Wb1 + bb1)  (K=384) ----
#pragma unroll 1
  for (int nt = 0; nt < 8; ++nt) {
    f32x8 acc = ZERO8;
#pragma unroll
    for (int kt = 0; kt < 12; ++kt)
      acc = wmma_bf16(ld_afrag(A2, S2, lane, kt * 32), ld_bfrag(pk + PK_WB1, kt * 8 + nt, lane), acc);
    float bias = bb1[nt * 16 + (lane & 15)];
#pragma unroll
    for (int j = 0; j < 8; ++j) acc[j] = fmaxf(acc[j] + bias, 0.0f);
    st_cfrag(W1, SW, lane, nt * 16, acc);
  }
  // ---- stage 5: w2 = relu(w1 @ Wb2 + bb2) ----
#pragma unroll 1
  for (int nt = 0; nt < 8; ++nt) {
    f32x8 acc = ZERO8;
#pragma unroll
    for (int kt = 0; kt < 4; ++kt)
      acc = wmma_bf16(ld_afrag(W1, SW, lane, kt * 32), ld_bfrag(pk + PK_WB2, kt * 8 + nt, lane), acc);
    float bias = bb2[nt * 16 + (lane & 15)];
#pragma unroll
    for (int j = 0; j < 8; ++j) acc[j] = fmaxf(acc[j] + bias, 0.0f);
    st_cfrag(W2, SW, lane, nt * 16, acc);
  }
  // ---- stage 6: logits = (w2 @ Wb3 + bb3) / sqrt(D); store + node atomicMax ----
  {
    f32x8 acc = ZERO8;
#pragma unroll
    for (int kt = 0; kt < 4; ++kt)
      acc = wmma_bf16(ld_afrag(W2, SW, lane, kt * 32), ld_bfrag(pk + PK_WB3, kt, lane), acc);
    int n = lane & 15, mb = (lane & 16) ? 8 : 0;
    if (n < 8) {
      float bias = bb3[n];
#pragma unroll
      for (int j = 0; j < 8; ++j) {
        float lg = (acc[j] + bias) * 0.17677669529663687f;   // 1/sqrt(32)
        int e = e0 + mb + j;
        wsLog[(size_t)e * 8 + n] = lg;
        atomicMax(&nodeMax[(size_t)center[e] * 8 + n], f2ord(lg));
      }
    }
  }
  // ---- stage 7: V1 = gelu(hE @ Wv1 + bv1)  (A = A2 cols 128..383) ----
#pragma unroll 1
  for (int nt = 0; nt < 8; ++nt) {
    f32x8 acc = ZERO8;
#pragma unroll
    for (int kt = 0; kt < 8; ++kt)
      acc = wmma_bf16(ld_afrag(A2, S2, lane, 128 + kt * 32), ld_bfrag(pk + PK_WV1, kt * 8 + nt, lane), acc);
    float bias = bv1[nt * 16 + (lane & 15)];
#pragma unroll
    for (int j = 0; j < 8; ++j) acc[j] = gelu_t(acc[j] + bias);
    st_cfrag(W1, SW, lane, nt * 16, acc);
  }
  // ---- stage 8: V2 = gelu(V1 @ Wv2 + bv2) ----
#pragma unroll 1
  for (int nt = 0; nt < 8; ++nt) {
    f32x8 acc = ZERO8;
#pragma unroll
    for (int kt = 0; kt < 4; ++kt)
      acc = wmma_bf16(ld_afrag(W1, SW, lane, kt * 32), ld_bfrag(pk + PK_WV2, kt * 8 + nt, lane), acc);
    float bias = bv2[nt * 16 + (lane & 15)];
#pragma unroll
    for (int j = 0; j < 8; ++j) acc[j] = gelu_t(acc[j] + bias);
    st_cfrag(W2, SW, lane, nt * 16, acc);
  }
  // ---- stage 9: V3 = V2 @ Wv3 + bv3 -> wsV (f32) ----
#pragma unroll 1
  for (int nt = 0; nt < 8; ++nt) {
    f32x8 acc = ZERO8;
#pragma unroll
    for (int kt = 0; kt < 4; ++kt)
      acc = wmma_bf16(ld_afrag(W2, SW, lane, kt * 32), ld_bfrag(pk + PK_WV3, kt * 8 + nt, lane), acc);
    int n = lane & 15, mb = (lane & 16) ? 8 : 0;
    float bias = bv3[nt * 16 + n];
#pragma unroll
    for (int j = 0; j < 8; ++j)
      wsV[(size_t)(e0 + mb + j) * 128 + nt * 16 + n] = acc[j] + bias;
  }
}

// ================= kernel 2: exp(logit - segmax) + node sums =================
__global__ void __launch_bounds__(256) softmax_kernel(
    float* __restrict__ wsLog, const unsigned int* __restrict__ nodeMax,
    float* __restrict__ nodeSum, const int* __restrict__ center) {
  int e = blockIdx.x * blockDim.x + threadIdx.x;
  if (e >= E_TOT) return;
  int c = center[e];
#pragma unroll
  for (int h = 0; h < 8; ++h) {
    float m = ord2f(nodeMax[(size_t)c * 8 + h]);
    float v = expf(wsLog[(size_t)e * 8 + h] - m);
    wsLog[(size_t)e * 8 + h] = v;
    atomicAdd(&nodeSum[(size_t)c * 8 + h], v);
  }
}

// ================= kernel 3: normalized scatter (h_agg, v_agg) =================
__global__ void __launch_bounds__(256) scatter_kernel(
    const float* __restrict__ expv, const float* __restrict__ nodeSum,
    const float* __restrict__ wsV, const float* __restrict__ wsVdt,
    const float* __restrict__ WvecV, const int* __restrict__ center,
    float* __restrict__ hagg, float* __restrict__ vagg) {
  int e = (blockIdx.x * blockDim.x + threadIdx.x) >> 5;
  int lane = threadIdx.x & 31;
  if (e >= E_TOT) return;
  int c = center[e];
  float at[8];
#pragma unroll
  for (int h = 0; h < 8; ++h)
    at[h] = expv[(size_t)e * 8 + h] / nodeSum[(size_t)c * 8 + h];
#pragma unroll
  for (int h = 0; h < 4; ++h) {
    int col = h * 32 + lane;
    atomicAdd(&hagg[(size_t)c * 128 + col], at[h] * wsV[(size_t)e * 128 + col]);
  }
  if (lane < 16) {
    int o = lane;
    float coef = at[4 + (o >> 2)];
    const float* vdt = wsVdt + (size_t)e * 48;
    float vx = 0.f, vy = 0.f, vz = 0.f;
#pragma unroll
    for (int i = 0; i < 16; ++i) {
      float w = WvecV[o * 16 + i];
      vx += w * vdt[i * 3 + 0];
      vy += w * vdt[i * 3 + 1];
      vz += w * vdt[i * 3 + 2];
    }
    atomicAdd(&vagg[(size_t)c * 48 + o * 3 + 0], coef * vx);
    atomicAdd(&vagg[(size_t)c * 48 + o * 3 + 1], coef * vy);
    atomicAdd(&vagg[(size_t)c * 48 + o * 3 + 2], coef * vz);
  }
}

// ================= kernel 4a: h_V_update = h_agg @ W_O (WMMA) =================
__global__ void __launch_bounds__(64) node_wo_kernel(
    const float* __restrict__ hagg, const unsigned short* __restrict__ pk,
    float* __restrict__ out) {
  __shared__ alignas(16) unsigned short sA[2][16 * SW];
  int wave = threadIdx.x >> 5, lane = threadIdx.x & 31;
  int r0 = (blockIdx.x * 2 + wave) * 16;
  unsigned short* A = sA[wave];
  {
    int row = lane >> 1, half = lane & 1;
    const float* src = hagg + (size_t)(r0 + row) * 128 + half * 64;
    unsigned short* dp = A + row * SW + half * 64;
#pragma unroll
    for (int i = 0; i < 16; ++i) {
      float4 v = ((const float4*)src)[i];
      *(ushort4*)(dp + i * 4) = make_ushort4(f2bf(v.x), f2bf(v.y), f2bf(v.z), f2bf(v.w));
    }
  }
#pragma unroll 1
  for (int nt = 0; nt < 8; ++nt) {
    f32x8 acc = ZERO8;
#pragma unroll
    for (int kt = 0; kt < 4; ++kt)
      acc = wmma_bf16(ld_afrag(A, SW, lane, kt * 32), ld_bfrag(pk + PK_WO, kt * 8 + nt, lane), acc);
    int n = lane & 15, mb = (lane & 16) ? 8 : 0;
#pragma unroll
    for (int j = 0; j < 8; ++j)
      out[(size_t)(r0 + mb + j) * 128 + nt * 16 + n] = acc[j];
  }
}

// ================= kernel 4b: vector merge / final projection =================
__global__ void __launch_bounds__(64) node_vec_kernel(
    const float* __restrict__ vagg, const float* __restrict__ hVvec,
    const float* __restrict__ Wmerge, const float* __restrict__ Wfin,
    const float* __restrict__ bfin, float* __restrict__ out) {
  __shared__ float sWm[512];
  __shared__ float sWf[2304];
  __shared__ float sbf[48];
  for (int i = threadIdx.x; i < 512; i += 64) sWm[i] = Wmerge[i];
  for (int i = threadIdx.x; i < 2304; i += 64) sWf[i] = Wfin[i];
  for (int i = threadIdx.x; i < 48; i += 64) sbf[i] = bfin[i];
  __syncthreads();
  int nidx = blockIdx.x * blockDim.x + threadIdx.x;
  if (nidx >= N_TOT) return;
  float vv[48], hv[48], v2[48];
#pragma unroll
  for (int i = 0; i < 48; ++i) vv[i] = vagg[(size_t)nidx * 48 + i];
#pragma unroll
  for (int i = 0; i < 48; ++i) hv[i] = hVvec[(size_t)nidx * 48 + i];
#pragma unroll
  for (int o = 0; o < 16; ++o) {
#pragma unroll
    for (int cc = 0; cc < 3; ++cc) {
      float s = 0.f;
#pragma unroll
      for (int i = 0; i < 16; ++i) s += sWm[o * 32 + i] * vv[i * 3 + cc];
#pragma unroll
      for (int i = 0; i < 16; ++i) s += sWm[o * 32 + 16 + i] * hv[i * 3 + cc];
      v2[o * 3 + cc] = s + vv[o * 3 + cc];
    }
  }
#pragma unroll 1
  for (int j = 0; j < 48; ++j) {
    float s = sbf[j];
#pragma unroll
    for (int i = 0; i < 48; ++i) s += v2[i] * sWf[i * 48 + j];
    out[(size_t)nidx * 48 + j] = s + hv[j];
  }
}

// ================= host launcher =================
extern "C" void kernel_launch(void* const* d_in, const int* in_sizes, int n_in,
                              void* d_out, int out_size, void* d_ws, size_t ws_size,
                              hipStream_t stream) {
  const float* hV      = (const float*)d_in[0];
  const float* hE      = (const float*)d_in[1];
  const int*   center  = (const int*)d_in[2];
  // d_in[3] = batch_id (unused by reference)
  const int*   dst     = (const int*)d_in[4];
  const float* frameR  = (const float*)d_in[5];
  const float* frameT  = (const float*)d_in[6];
  const float* hVvec   = (const float*)d_in[7];
  const float* Wevf    = (const float*)d_in[8];
  const float* bnG     = (const float*)d_in[9];
  const float* bnB     = (const float*)d_in[10];
  const float* bnM     = (const float*)d_in[11];
  const float* bnV     = (const float*)d_in[12];
  const float* Wred    = (const float*)d_in[13];
  const float* Wb1     = (const float*)d_in[14];
  const float* bb1     = (const float*)d_in[15];
  const float* Wb2     = (const float*)d_in[16];
  const float* bb2     = (const float*)d_in[17];
  const float* Wb3     = (const float*)d_in[18];
  const float* bb3     = (const float*)d_in[19];
  const float* Wv1     = (const float*)d_in[20];
  const float* bv1     = (const float*)d_in[21];
  const float* Wv2     = (const float*)d_in[22];
  const float* bv2     = (const float*)d_in[23];
  const float* Wv3     = (const float*)d_in[24];
  const float* bv3     = (const float*)d_in[25];
  const float* WO      = (const float*)d_in[26];
  const float* WvecV   = (const float*)d_in[27];
  const float* Wmerge  = (const float*)d_in[28];
  const float* Wfin    = (const float*)d_in[29];
  const float* bfin    = (const float*)d_in[30];

  char* ws = (char*)d_ws;
  unsigned short* pk      = (unsigned short*)(ws + OFF_PACK);
  float*          wsVdt   = (float*)(ws + OFF_VDT);
  float*          wsLog   = (float*)(ws + OFF_LOG);
  float*          wsV     = (float*)(ws + OFF_V);
  unsigned int*   nodeMax = (unsigned int*)(ws + OFF_NMAX);
  float*          nodeSum = (float*)(ws + OFF_NSUM);
  float*          hagg    = (float*)(ws + OFF_HAGG);
  float*          vagg    = (float*)(ws + OFF_VAGG);

  float* outHV   = (float*)d_out;                        // N*128
  float* outVec  = (float*)d_out + (size_t)N_TOT * 128;  // N*48
  float* outFeat = outVec + (size_t)N_TOT * 48;          // E*64

  // 0) pack weights (bf16, fragment order)
  pack_kernel<<<(PK_FRAGS * 32 + 255) / 256, 256, 0, stream>>>(
      Wred, Wb1, Wb2, Wb3, Wv1, Wv2, Wv3, WO, pk);
  // 0b) zero node accumulators (contiguous region)
  hipMemsetAsync(ws + OFF_NMAX, 0, ZCLEAR_BYTES, stream);
  // 1) edge fused WMMA chain (16 edges per wave, 2 waves per block)
  edge_kernel<<<E_TOT / 32, 64, 0, stream>>>(
      hV, hE, center, dst, frameR, frameT, hVvec, Wevf, bnG, bnB, bnM, bnV,
      bb1, bb2, bb3, bv1, bv2, bv3, pk, wsVdt, wsLog, wsV, nodeMax, outFeat);
  // 2) segment softmax: exp + sums
  softmax_kernel<<<E_TOT / 256, 256, 0, stream>>>(wsLog, nodeMax, nodeSum, center);
  // 3) weighted scatter into node accumulators
  scatter_kernel<<<E_TOT / 8, 256, 0, stream>>>(
      wsLog, nodeSum, wsV, wsVdt, WvecV, center, hagg, vagg);
  // 4a) h_V_update = h_agg @ W_O
  node_wo_kernel<<<N_TOT / 32, 64, 0, stream>>>(hagg, pk, outHV);
  // 4b) vector merge + final projection + residual
  node_vec_kernel<<<N_TOT / 64, 64, 0, stream>>>(vagg, hVvec, Wmerge, Wfin, bfin, outVec);
}
